// VectorQuantizer_85779086835953
// MI455X (gfx1250) — compile-verified, matrix operand order checked by reading, not run on hardware
//
#include <hip/hip_runtime.h>
#include <hip/hip_bf16.h>
#include <math.h>

// ---------- problem constants ----------
#define NROWS   16384        // 32*512 query vectors
#define NCODES  8192         // codebook size
#define DIM     512          // embedding dim
#define BETA    0.25f

// output layout (floats, concatenated in reference return order)
#define O_LOSS  ((size_t)0)
#define O_ZQ    ((size_t)1)
#define O_PERP  ((size_t)8388609)
#define O_ENC   ((size_t)8388610)
#define O_IDX   ((size_t)142606338)

typedef __bf16 bf16x16 __attribute__((ext_vector_type(16)));
typedef float  v8f     __attribute__((ext_vector_type(8)));
typedef unsigned int v4u __attribute__((ext_vector_type(4)));

union Frag { bf16x16 v; v4u u[2]; };

__device__ __forceinline__ unsigned short f2bf(float f) {
    unsigned int u = __float_as_uint(f);
    unsigned int r = ((u >> 16) & 1u) + 0x7fffu;   // round-to-nearest-even
    return (unsigned short)((u + r) >> 16);
}

// ---------- k0: zero one-hot output region + counters ----------
__global__ void vq_zero(float* __restrict__ out, int* __restrict__ cnt) {
    size_t gid = (size_t)blockIdx.x * blockDim.x + threadIdx.x;   // 33,554,432 threads
    float* enc = out + O_ENC;
#pragma unroll
    for (int i = 0; i < 4; ++i)
        enc[gid + (size_t)i * 33554432u] = 0.0f;
    if (gid < NCODES) cnt[gid] = 0;
}

// ---------- k1: convert z and emb to bf16 ----------
__global__ void vq_convert(const float* __restrict__ z, const float* __restrict__ emb,
                           unsigned short* __restrict__ zb, unsigned short* __restrict__ eb) {
    size_t gid = (size_t)blockIdx.x * blockDim.x + threadIdx.x;   // 12,582,912 threads
    const size_t nz = (size_t)NROWS * DIM;
    if (gid < nz)      zb[gid]        = f2bf(z[gid]);
    else               eb[gid - nz]   = f2bf(emb[gid - nz]);
}

// ---------- k1b: ||e_j||^2 per code ----------
__global__ void vq_e2(const float* __restrict__ emb, float* __restrict__ e2) {
    int j = blockIdx.x, t = threadIdx.x;
    const float* er = emb + (size_t)j * DIM;
    float a = er[t], b = er[t + 256];
    float s = a * a + b * b;
    __shared__ float red[256];
    red[t] = s; __syncthreads();
    for (int o = 128; o > 0; o >>= 1) { if (t < o) red[t] += red[t + o]; __syncthreads(); }
    if (t == 0) e2[j] = red[0];
}

// async global -> LDS copy of one 16-byte chunk per lane (bypasses VGPRs, ASYNCcnt)
__device__ __forceinline__ void async_cp16(unsigned lds_off, unsigned long long gaddr) {
    asm volatile("global_load_async_to_lds_b128 %0, %1, off"
                 :: "v"(lds_off), "v"(gaddr) : "memory");
}
__device__ __forceinline__ void async_wait0() {
    asm volatile("s_wait_asynccnt 0x0" ::: "memory");
}

// ---------- k2: fused bf16-WMMA GEMM (-2 z.e^T + ||e||^2) with argmin epilogue ----------
__global__ void __launch_bounds__(256)
vq_gemm_argmin(const unsigned short* __restrict__ zb, const unsigned short* __restrict__ eb,
               const float* __restrict__ e2, int* __restrict__ idxw) {
    __shared__ v4u Bsm[2][1024];                  // 2 x (16 codes x 512 bf16) = 32 KB

    const int t    = threadIdx.x;
    const int lane = t & 31;
    const int wave = t >> 5;
    const int h    = lane >> 4;                   // half-wave select (K grouping)
    const int nl   = lane & 15;                   // M (for A) / N (for B,C)
    const int rowBase = blockIdx.x * 128 + wave * 16;

    // Preload this wave's A fragments for all 16 K-tiles (row rowBase+nl, full K=512).
    const v4u* arow = (const v4u*)(zb + (size_t)(rowBase + nl) * DIM);
    Frag a[16];
#pragma unroll
    for (int kt = 0; kt < 16; ++kt) {
        a[kt].u[0] = arow[4 * kt + h];            // K = 32*kt + 8*h  .. +7
        a[kt].u[1] = arow[4 * kt + h + 2];        // K = 32*kt + 16 + 8*h .. +7
    }

    float best[8]; int bidx[8];
#pragma unroll
    for (int v = 0; v < 8; ++v) { best[v] = 3.4e38f; bidx[v] = 0; }

    // per-thread staging addresses: thread t copies bytes [t*16 + i*4096) of each tile
    const unsigned lds0 = (unsigned)(unsigned long long)(const void*)&Bsm[0][t];
    const unsigned lds1 = (unsigned)(unsigned long long)(const void*)&Bsm[1][t];
    const unsigned long long gbase = (unsigned long long)(const void*)eb + (unsigned long long)t * 16u;

    // stage B tile 0 asynchronously (global -> LDS direct)
#pragma unroll
    for (int i = 0; i < 4; ++i)
        async_cp16(lds0 + i * 4096u, gbase + (unsigned long long)i * 4096u);
    async_wait0();
    __syncthreads();

    for (int jt = 0; jt < 512; ++jt) {
        const int buf = jt & 1;
        if (jt + 1 < 512) {                       // async prefetch next 16-code tile
            const unsigned dst = buf ? lds0 : lds1;
            const unsigned long long src = gbase + (unsigned long long)(jt + 1) * (16u * DIM * 2u);
#pragma unroll
            for (int i = 0; i < 4; ++i)
                async_cp16(dst + i * 4096u, src + (unsigned long long)i * 4096u);
        }

        const v4u* brow = (const v4u*)&Bsm[buf][nl * 64];   // code row nl of tile
        v8f c = {};
#pragma unroll
        for (int kt = 0; kt < 16; ++kt) {
            Frag b;
            b.u[0] = brow[4 * kt + h];
            b.u[1] = brow[4 * kt + h + 2];
            c = __builtin_amdgcn_wmma_f32_16x16x32_bf16(
                    false, a[kt].v, false, b.v, (short)0, c, false, false);
        }

        // epilogue: d = ||e_j||^2 - 2 * (z.e_j); running argmin per accumulator row
        const int j = jt * 16 + nl;
        const float e2j = e2[j];
#pragma unroll
        for (int v = 0; v < 8; ++v) {
            float d = fmaf(-2.0f, c[v], e2j);
            if (d < best[v]) { best[v] = d; bidx[v] = j; }
        }

        async_wait0();        // this wave's staged bytes are in LDS
        __syncthreads();      // publish to all waves before next compute
    }

    // reduce across the 16 lanes of each half (C layout: N = lane&15)
#pragma unroll
    for (int m = 1; m < 16; m <<= 1) {
#pragma unroll
        for (int v = 0; v < 8; ++v) {
            float ov = __shfl_xor(best[v], m, 32);
            int   oi = __shfl_xor(bidx[v], m, 32);
            if (ov < best[v] || (ov == best[v] && oi < bidx[v])) { best[v] = ov; bidx[v] = oi; }
        }
    }
    if (nl == 0) {
#pragma unroll
        for (int v = 0; v < 8; ++v)
            idxw[rowBase + v + 8 * h] = bidx[v];   // M = vgpr + 8*(lane>=16)
    }
}

// ---------- k3: per-row gather z_q, squared error, one-hot scatter, counts ----------
__global__ void vq_row_finalize(const float* __restrict__ z, const float* __restrict__ emb,
                                const int* __restrict__ idxw, int* __restrict__ cnt,
                                float* __restrict__ row_loss, float* __restrict__ out) {
    int row = blockIdx.x, t = threadIdx.x;
    int j = idxw[row];
    const float* er = emb + (size_t)j * DIM;
    const float* zr = z   + (size_t)row * DIM;
    float* zq = out + O_ZQ + (size_t)row * DIM;   // z_q_st == z_q numerically
    float acc = 0.0f;
    for (int k = t; k < DIM; k += 256) {
        float e = er[k];
        zq[k] = e;
        float d = e - zr[k];
        acc += d * d;
    }
    __shared__ float red[256];
    red[t] = acc; __syncthreads();
    for (int o = 128; o > 0; o >>= 1) { if (t < o) red[t] += red[t + o]; __syncthreads(); }
    if (t == 0) {
        row_loss[row] = red[0];
        atomicAdd(&cnt[j], 1);
        out[O_IDX + row] = (float)j;
        out[O_ENC + (size_t)row * NCODES + j] = 1.0f;
    }
}

// ---------- k4: deterministic scalar reductions (loss, perplexity) ----------
__global__ void vq_finalize_scalars(const int* __restrict__ cnt,
                                    const float* __restrict__ row_loss,
                                    float* __restrict__ out) {
    __shared__ float s1[256], s2[256];
    int t = threadIdx.x;
    float ls = 0.0f;
    for (int r = t; r < NROWS; r += 256) ls += row_loss[r];
    float ps = 0.0f;
    for (int j = t; j < NCODES; j += 256) {
        float p = (float)cnt[j] * (1.0f / (float)NROWS);
        ps += p * logf(p + 1e-10f);
    }
    s1[t] = ls; s2[t] = ps; __syncthreads();
    for (int o = 128; o > 0; o >>= 1) {
        if (t < o) { s1[t] += s1[t + o]; s2[t] += s2[t + o]; }
        __syncthreads();
    }
    if (t == 0) {
        out[O_LOSS] = (1.0f + BETA) * s1[0] / (float)((size_t)NROWS * DIM);
        out[O_PERP] = expf(-s2[0]);
    }
}

extern "C" void kernel_launch(void* const* d_in, const int* in_sizes, int n_in,
                              void* d_out, int out_size, void* d_ws, size_t ws_size,
                              hipStream_t stream) {
    (void)in_sizes; (void)n_in; (void)out_size; (void)ws_size;
    const float* z   = (const float*)d_in[0];     // [16384, 512]
    const float* emb = (const float*)d_in[1];     // [8192, 512]
    float* out = (float*)d_out;

    // workspace partition (all offsets 16B-aligned)
    unsigned short* zb = (unsigned short*)d_ws;                       // 16 MB
    unsigned short* eb = zb + (size_t)NROWS * DIM;                    //  8 MB
    float* e2          = (float*)(eb + (size_t)NCODES * DIM);         // 32 KB
    int*   idxw        = (int*)(e2 + NCODES);                         // 64 KB
    int*   cnt         = idxw + NROWS;                                // 32 KB
    float* row_loss    = (float*)(cnt + NCODES);                      // 64 KB

    vq_zero<<<131072, 256, 0, stream>>>(out, cnt);
    vq_convert<<<49152, 256, 0, stream>>>(z, emb, zb, eb);
    vq_e2<<<NCODES, 256, 0, stream>>>(emb, e2);
    vq_gemm_argmin<<<NROWS / 128, 256, 0, stream>>>(zb, eb, e2, idxw);
    vq_row_finalize<<<NROWS, 256, 0, stream>>>(z, emb, idxw, cnt, row_loss, out);
    vq_finalize_scalars<<<1, 256, 0, stream>>>(cnt, row_loss, out);
}
